// PhaseAwareAttention_4329327035082
// MI455X (gfx1250) — matrix operand order repaired and checked by reading, not compiled, not run on hardware
//
#include <hip/hip_runtime.h>

typedef __attribute__((ext_vector_type(16))) _Float16 v16h;
typedef __attribute__((ext_vector_type(8)))  float    v8f;

#define HID 128

// ---------------------------------------------------------------------------
// Stage kernels: fuse the dead-softmax-eliminated weight chain
//   M = Wo @ W_mo @ Wiv @ Wv          (each 128x128, row-major)
//   b = Wo@(W_mo@(Wiv@bv + biv) + b_mo) + bo
// Tiny one-shot work; plain VALU kernels.
// ---------------------------------------------------------------------------

// C[i][j] = sum_k A[i][k]*B[k][j], 128x128 row-major. Launch <<<64,256>>>.
__global__ void fuse_gemm128(const float* __restrict__ A,
                             const float* __restrict__ B,
                             float* __restrict__ C) {
  int idx = blockIdx.x * 256 + threadIdx.x;
  int i = idx >> 7, j = idx & 127;
  float s = 0.f;
#pragma unroll 8
  for (int k = 0; k < 128; ++k) s += A[i * 128 + k] * B[k * 128 + j];
  C[i * 128 + j] = s;
}

// Swizzle fused M into per-lane WMMA f16 B-fragment order + compute fused bias.
// B order used here (contiguous K, same as A): lane l -> column N = l&15,
// K = kc*32 + (l>>4)*16 + i, element i of the per-lane v16h. B[k][n] = M[n][k].
// Launch <<<1,256>>>.
__global__ void fuse_pack(const float* __restrict__ M,
                          const float* __restrict__ Wiv,
                          const float* __restrict__ bv,
                          const float* __restrict__ biv,
                          const float* __restrict__ Wmo,
                          const float* __restrict__ bmo,
                          const float* __restrict__ Wo,
                          const float* __restrict__ bo,
                          _Float16* __restrict__ Bpack,
                          float* __restrict__ biasOut) {
  int t = threadIdx.x;
  for (int idx = t; idx < 128 * 128; idx += 256) {
    int n = idx >> 7, k = idx & 127;
    int nt = n >> 4, lnN = n & 15;
    int kc = k >> 5, kr = k & 31;
    int lane = ((kr >> 4) << 4) + lnN;
    int i16 = kr & 15;
    Bpack[((((nt << 2) + kc) * 32 + lane) << 4) + i16] = (_Float16)M[idx];
  }
  __shared__ float c1[128], c2[128];
  if (t < 128) {
    float s = biv[t];
    for (int k = 0; k < 128; ++k) s += Wiv[t * 128 + k] * bv[k];
    c1[t] = s;
  }
  __syncthreads();
  if (t < 128) {
    float s = bmo[t];
    for (int k = 0; k < 128; ++k) s += Wmo[t * 128 + k] * c1[k];
    c2[t] = s;
  }
  __syncthreads();
  if (t < 128) {
    float s = bo[t];
    for (int k = 0; k < 128; ++k) s += Wo[t * 128 + k] * c2[k];
    biasOut[t] = s;
  }
}

// ---------------------------------------------------------------------------
// Main streaming kernel: out = x + 0.5*(x @ M.T + b), x = edge_attr [E,128].
// One wave32 per 16-row tile; v_wmma_f32_16x16x32_f16, f32 accumulate.
// Weight fragments staged once per workgroup in LDS (32 KB).
// ---------------------------------------------------------------------------
__global__ void __launch_bounds__(256) edge_mlp_wmma(
    const float* __restrict__ x,
    const _Float16* __restrict__ Bpack,   // [8 nt][4 kc][32 lane][16] f16
    const float* __restrict__ bias,
    float* __restrict__ out,
    int ntiles, long E) {
  __shared__ __align__(32) _Float16 lb[8 * 4 * 32 * 16];  // 32 KB
  {
    const uint4* s = (const uint4*)Bpack;
    uint4* d = (uint4*)lb;
#pragma unroll
    for (int i = 0; i < 8; ++i)
      d[threadIdx.x + i * 256] = s[threadIdx.x + i * 256];
  }
  __syncthreads();

  const int wave = threadIdx.x >> 5;
  const int lane = threadIdx.x & 31;
  const int tile = blockIdx.x * 8 + wave;
  if (tile >= ntiles) return;               // wave-uniform: EXEC all-1s for WMMA

  const int half = lane >> 4;
  const int ln = lane & 15;
  const long base = (long)tile * 16;

  // ---- A fragments (16x32 f16): lane = row M, K contiguous per lane half ----
  long arow = base + ln;
  if (arow >= E) arow = E - 1;              // tail clamp (reads only)
  const float* xp = x + arow * HID;

  v16h a[4];
#pragma unroll
  for (int kc = 0; kc < 4; ++kc) {
    const float* p = xp + kc * 32 + half * 16;
#pragma unroll
    for (int i = 0; i < 8; ++i) {
      a[kc][i]     = (_Float16)p[i];        // K = kc*32 + half*16 + i
      a[kc][8 + i] = (_Float16)p[8 + i];    // K = kc*32 + half*16 + 8 + i
    }
  }

  // ---- 8 n-tiles x 4 k-chunks of v_wmma_f32_16x16x32_f16 ----
  v8f acc[8];
#pragma unroll
  for (int nt = 0; nt < 8; ++nt) {
    v8f c = {0.f, 0.f, 0.f, 0.f, 0.f, 0.f, 0.f, 0.f};
#pragma unroll
    for (int kc = 0; kc < 4; ++kc) {
      v16h b = *(const v16h*)(lb + ((((nt << 2) + kc) * 32 + lane) << 4));
      c = __builtin_amdgcn_wmma_f32_16x16x32_f16(
          /*neg_a=*/false, a[kc], /*neg_b=*/false, b,
          /*c_mod=*/(short)0, c, /*reuse_a=*/false, /*reuse_b=*/false);
    }
    acc[nt] = c;
  }

  // ---- residual + bias + store (D layout: lane=col, VGPR j = row j+8*half) ----
#pragma unroll
  for (int nt = 0; nt < 8; ++nt) {
    const int col = nt * 16 + ln;
    const float bc = bias[col];
#pragma unroll
    for (int j = 0; j < 8; ++j) {
      long r = base + half * 8 + j;
      if (r < E) {
        long idx = r * HID + col;
        out[idx] = x[idx] + 0.5f * (acc[nt][j] + bc);
      }
    }
  }
}

// ---------------------------------------------------------------------------
extern "C" void kernel_launch(void* const* d_in, const int* in_sizes, int n_in,
                              void* d_out, int out_size, void* d_ws, size_t ws_size,
                              hipStream_t stream) {
  (void)n_in; (void)out_size; (void)ws_size;
  const float* x   = (const float*)d_in[0];   // edge_attr [E,128]
  const float* Wv  = (const float*)d_in[7];
  const float* bv  = (const float*)d_in[8];
  const float* Win = (const float*)d_in[9];   // [384,128]
  const float* bin = (const float*)d_in[10];
  const float* Wmo = (const float*)d_in[11];
  const float* bmo = (const float*)d_in[12];
  const float* Wo  = (const float*)d_in[13];
  const float* bo  = (const float*)d_in[14];
  float* out = (float*)d_out;

  char* ws = (char*)d_ws;
  _Float16* Bpack = (_Float16*)(ws);          // 32 KB packed f16 fragments
  float* biasF    = (float*)(ws + 32768);     // 512 B fused bias
  float* T1       = (float*)(ws + 65536);     // 64 KB
  float* T2       = (float*)(ws + 131072);    // 64 KB
  float* Mf       = T1;                       // reuse T1 for final M

  const float* Wiv = Win + 2 * HID * HID;     // value slice of packed in-proj
  const float* biv = bin + 2 * HID;

  // Weight fusion (softmax over seq_len==1 is identically 1 -> q/k paths dead)
  fuse_gemm128<<<64, 256, 0, stream>>>(Wiv, Wv, T1);    // T1 = Wiv@Wv
  fuse_gemm128<<<64, 256, 0, stream>>>(Wmo, T1, T2);    // T2 = Wmo@T1
  fuse_gemm128<<<64, 256, 0, stream>>>(Wo,  T2, Mf);    // M  = Wo@T2
  fuse_pack<<<1, 256, 0, stream>>>(Mf, Wiv, bv, biv, Wmo, bmo, Wo, bo,
                                   Bpack, biasF);

  long E = (long)in_sizes[0] / HID;
  int ntiles = (int)((E + 15) / 16);
  int blocks = (ntiles + 7) / 8;
  edge_mlp_wmma<<<blocks, 256, 0, stream>>>(x, Bpack, biasF, out, ntiles, E);
}
